// AssociativeMemory_7344394076209
// MI455X (gfx1250) — matrix operand order repaired and checked by reading, not compiled, not run on hardware
//
#include <hip/hip_runtime.h>
#include <hip/hip_bf16.h>
#include <math.h>

#define P 8192
#define ITERS 10

typedef __attribute__((ext_vector_type(16))) __bf16        v16bf;
typedef __attribute__((ext_vector_type(8)))  float         v8f;
typedef __attribute__((ext_vector_type(4)))  float         v4f;
typedef __attribute__((ext_vector_type(4)))  unsigned int  v4u;
typedef __attribute__((ext_vector_type(4)))  unsigned short v4us;

union ABReg { v4u q[2]; v16bf v; };

__device__ __forceinline__ unsigned short f32_to_bf16(float f) {
  union { float f; unsigned int u; } x; x.f = f;
  unsigned int r = x.u + 0x7FFFu + ((x.u >> 16) & 1u);  // round-to-nearest-even
  return (unsigned short)(r >> 16);
}

// fp32 -> bf16 conversion, 4 elements/thread, grid-stride.
// Non-temporal loads: the fp32 source is never read again, keep L2 for the bf16 copy.
__global__ void hopfield_convert_bf16(const float* __restrict__ src,
                                      unsigned short* __restrict__ dst,
                                      long long n) {
  long long i = (long long)blockIdx.x * blockDim.x + threadIdx.x;
  long long stride = (long long)gridDim.x * blockDim.x;
  for (long long e = i * 4; e < n; e += stride * 4) {
    v4f f = __builtin_nontemporal_load((const v4f*)(src + e));
    v4us o;
    o[0] = f32_to_bf16(f[0]); o[1] = f32_to_bf16(f[1]);
    o[2] = f32_to_bf16(f[2]); o[3] = f32_to_bf16(f[3]);
    *(v4us*)(dst + e) = o;
  }
}

// 4 waves per block cooperate on one 16-row tile; each wave reduces a quarter
// of K via v_wmma_f32_16x16x32_bf16 (x broadcast into all 16 B columns, so D
// column 0 is the exact matvec), then partials are combined through LDS.
__global__ __launch_bounds__(128) void hopfield_gemv_wmma(
    const unsigned short* __restrict__ Wb,   // P x P bf16, row-major
    const unsigned short* __restrict__ xb,   // P bf16 (current state)
    unsigned short* __restrict__ yb,         // P bf16 (next state)
    float* __restrict__ yf,                  // P f32 (final output)
    int writeF) {
  __shared__ float part[4][16];

  const int lane = threadIdx.x & 31;
  const int wave = threadIdx.x >> 5;
  const int tile = blockIdx.x;             // 0..P/16-1
  const int row  = tile * 16 + (lane & 15);              // A: M = lane%16
  const size_t rowbase = (size_t)row * P;
  const int aoff = (lane < 16) ? 0 : 8;    // A K-offset per lane group (elements)
  const int boff = (lane < 16) ? 0 : 4;    // B dword offset per lane group

  const unsigned int* xw = (const unsigned int*)xb;      // bf16 pairs

  const int kbeg = wave * (P / 4);
  const int kend = kbeg + (P / 4);

  v8f c0 = {}; v8f c1 = {};
  for (int k = kbeg; k < kend; k += 64) {
    ABReg a0, a1, b0, b1;
    const unsigned short* ap = Wb + rowbase + k + aoff;
    a0.q[0] = *(const v4u*)(ap);           // K = base+0..7   (this lane group)
    a0.q[1] = *(const v4u*)(ap + 16);      // K = base+16..23
    a1.q[0] = *(const v4u*)(ap + 32);
    a1.q[1] = *(const v4u*)(ap + 48);
    const unsigned int* bp = xw + (k >> 1) + boff;
    b0.q[0] = *(const v4u*)(bp);           // K = base+0..7 (+8 for hi lanes), then +16..23
    b0.q[1] = *(const v4u*)(bp + 8);
    b1.q[0] = *(const v4u*)(bp + 16);
    b1.q[1] = *(const v4u*)(bp + 24);
    c0 = __builtin_amdgcn_wmma_f32_16x16x32_bf16(false, a0.v, false, b0.v,
                                                 (short)0, c0, false, false);
    c1 = __builtin_amdgcn_wmma_f32_16x16x32_bf16(false, a1.v, false, b1.v,
                                                 (short)0, c1, false, false);
  }

  // D column 0: lane 0 holds M=0..7 (VGPR r -> M=r), lane 16 holds M=8..15.
  if (lane == 0 || lane == 16) {
    const int mb = (lane == 16) ? 8 : 0;
#pragma unroll
    for (int r = 0; r < 8; ++r) part[wave][mb + r] = c0[r] + c1[r];
  }
  __syncthreads();

  if (threadIdx.x < 16) {
    const int m = tile * 16 + threadIdx.x;
    float s = part[0][threadIdx.x] + part[1][threadIdx.x] +
              part[2][threadIdx.x] + part[3][threadIdx.x];
    float t = tanhf(s);
    yb[m] = f32_to_bf16(t);
    if (writeF) yf[m] = t;
  }
}

// Fallback (workspace too small for bf16 W): plain fp32 GEMV, one block per row.
__global__ void hopfield_gemv_f32(const float* __restrict__ W,
                                  const float* __restrict__ x,
                                  float* __restrict__ y) {
  __shared__ float red[256];
  const float* wr = W + (size_t)blockIdx.x * P;
  float s = 0.f;
  for (int k = threadIdx.x * 4; k < P; k += 256 * 4) {
    v4f w = *(const v4f*)(wr + k);
    v4f v = *(const v4f*)(x + k);
    s += w[0] * v[0] + w[1] * v[1] + w[2] * v[2] + w[3] * v[3];
  }
  red[threadIdx.x] = s;
  __syncthreads();
  for (int off = 128; off > 0; off >>= 1) {
    if (threadIdx.x < off) red[threadIdx.x] += red[threadIdx.x + off];
    __syncthreads();
  }
  if (threadIdx.x == 0) y[blockIdx.x] = tanhf(red[0]);
}

extern "C" void kernel_launch(void* const* d_in, const int* in_sizes, int n_in,
                              void* d_out, int out_size, void* d_ws, size_t ws_size,
                              hipStream_t stream) {
  const float* x = (const float*)d_in[0];     // (P,)
  const float* W = (const float*)d_in[1];     // (P, P)
  float* out = (float*)d_out;                 // (P,) f32

  const size_t needW = (size_t)P * P * sizeof(unsigned short);   // 128 MiB
  const size_t needS = 2 * (size_t)P * sizeof(unsigned short);   // ping-pong bf16 states

  if (ws_size >= needW + needS) {
    unsigned short* Wb = (unsigned short*)d_ws;
    unsigned short* s0 = (unsigned short*)((char*)d_ws + needW);
    unsigned short* s1 = s0 + P;

    hopfield_convert_bf16<<<16384, 256, 0, stream>>>(W, Wb, (long long)P * P);
    hopfield_convert_bf16<<<8, 256, 0, stream>>>(x, s0, (long long)P);

    unsigned short* cur = s0;
    unsigned short* nxt = s1;
    for (int it = 0; it < ITERS; ++it) {
      const int last = (it == ITERS - 1) ? 1 : 0;
      hopfield_gemv_wmma<<<P / 16, 128, 0, stream>>>(Wb, cur, nxt, out, last);
      unsigned short* t = cur; cur = nxt; nxt = t;
    }
  } else {
    // fp32 fallback: needs only 2*P floats of scratch
    float* s0 = (float*)d_ws;
    float* s1 = s0 + P;
    hipMemcpyAsync(s0, x, P * sizeof(float), hipMemcpyDeviceToDevice, stream);
    float* cur = s0;
    float* nxt = s1;
    for (int it = 0; it < ITERS; ++it) {
      float* dst = (it == ITERS - 1) ? out : nxt;
      hopfield_gemv_f32<<<P, 256, 0, stream>>>(W, cur, dst);
      float* t = cur; cur = nxt; nxt = t;
    }
  }
}
